// MLP_3977139716762
// MI455X (gfx1250) — matrix operand order repaired and checked by reading, not compiled, not run on hardware
//
#include <hip/hip_runtime.h>
#include <hip/hip_bf16.h>
#include <type_traits>

// ---------------------------------------------------------------------------
// MoE top-2 MLP for MI455X (gfx1250).
//   T=8192 tokens, H=1024, F=4096, E=8, TOPK=2.
// Router+compaction -> fused per-expert GEMM (bf16 WMMA), M=48 token tiles,
// optional one-shot fp32->bf16 weight conversion in workspace,
// async global->LDS staging of the h tile (ASYNCcnt path).
// ---------------------------------------------------------------------------

typedef __attribute__((ext_vector_type(16))) __bf16 v16bf;
typedef __attribute__((ext_vector_type(8)))  __bf16 v8bf;
typedef __attribute__((ext_vector_type(4)))  __bf16 v4bf;
typedef __attribute__((ext_vector_type(8)))  float  v8f;
typedef int v4i __attribute__((vector_size(4 * sizeof(int))));

#define T_TOK   8192
#define H_DIM   1024
#define F_DIM   4096
#define NEXP    8
#define MTILE   48                 // tokens per workgroup = 3 WMMA M-tiles
#define MT_N    (MTILE / 16)       // 3
#define FB      256                // F-chunk per fused iteration
#define NWAVES  16
#define NTHREADS (NWAVES * 32)     // 512
#define HS      (H_DIM + 8)        // LDS row stride (16B-aligned rows, bank skew)
#define FS      (FB + 8)

#if defined(__gfx1250__) && __has_builtin(__builtin_amdgcn_global_load_async_to_lds_b128) && __has_builtin(__builtin_amdgcn_s_wait_asynccnt)
#define USE_ASYNC_LDS 1
typedef __attribute__((address_space(1))) v4i g_v4i;   // global int4
typedef __attribute__((address_space(3))) v4i l_v4i;   // LDS int4
#endif

__device__ inline float fast_rcp(float x) {
#if defined(__gfx1250__) && __has_builtin(__builtin_amdgcn_rcpf)
    return __builtin_amdgcn_rcpf(x);
#else
    return 1.f / x;
#endif
}

// A-fragment (16x32 bf16, MxK) from LDS, contiguous K order (matches B):
//   lane<16 : M=lane,    elems 0..15 -> K=0..15
//   lane>=16: M=lane-16, elems 0..15 -> K=16..31
__device__ inline v16bf load_a_frag_lds(const __bf16* base, int stride, int k0, int lane) {
    const int m  = lane & 15;
    const int kh = (lane >> 4) * 16;
    const __bf16* p = base + (size_t)m * stride + k0 + kh;
    v8bf lo = *(const v8bf*)(p);
    v8bf hi = *(const v8bf*)(p + 8);
    v16bf a;
#pragma unroll
    for (int i = 0; i < 8; ++i) { a[i] = lo[i]; a[i + 8] = hi[i]; }
    return a;
}

// B-fragment (32x16 bf16, KxN), B[k][n] = W[n0+n][k0+k] (GEMM with W^T).
// lanes 0-15 hold K=0..15, lanes 16-31 hold K=16..31; column N = lane%16.
__device__ inline v16bf load_b_frag(const float* __restrict__ W, size_t rstride,
                                    int n0, int k0, int lane) {
    const float4* p = (const float4*)(W + (size_t)(n0 + (lane & 15)) * rstride
                                        + k0 + (lane >> 4) * 16);
    v16bf b;
#pragma unroll
    for (int q = 0; q < 4; ++q) {
        float4 v = p[q];
        b[q * 4 + 0] = (__bf16)v.x;  b[q * 4 + 1] = (__bf16)v.y;
        b[q * 4 + 2] = (__bf16)v.z;  b[q * 4 + 3] = (__bf16)v.w;
    }
    return b;
}

__device__ inline v16bf load_b_frag(const __bf16* __restrict__ W, size_t rstride,
                                    int n0, int k0, int lane) {
    const __bf16* p = W + (size_t)(n0 + (lane & 15)) * rstride + k0 + (lane >> 4) * 16;
    v8bf lo = *(const v8bf*)(p);       // 16 consecutive K -> two 16B loads
    v8bf hi = *(const v8bf*)(p + 8);
    v16bf b;
#pragma unroll
    for (int i = 0; i < 8; ++i) { b[i] = lo[i]; b[i + 8] = hi[i]; }
    return b;
}

// ---------------------------------------------------------------------------
// fp32 -> bf16 bulk conversion (grid-stride, float4 in / 4x bf16 out).
// ---------------------------------------------------------------------------
__global__ __launch_bounds__(256) void cvt_f32_bf16_kernel(
    const float* __restrict__ src, __bf16* __restrict__ dst, long long n)
{
    long long i = (long long)blockIdx.x * blockDim.x + threadIdx.x;
    long long stride = (long long)gridDim.x * blockDim.x;
    for (long long idx = i * 4; idx < n; idx += stride * 4) {
        float4 v = *(const float4*)(src + idx);
        v4bf o;
        o[0] = (__bf16)v.x; o[1] = (__bf16)v.y; o[2] = (__bf16)v.z; o[3] = (__bf16)v.w;
        *(v4bf*)(dst + idx) = o;
    }
}

// ---------------------------------------------------------------------------
// Router: one wave32 per token; E=8 dots over H, fp32 softmax, top-2,
// renormalize, compact into per-expert gather lists.
// ---------------------------------------------------------------------------
__global__ __launch_bounds__(256) void moe_router_kernel(
    const float* __restrict__ x, const float* __restrict__ Wr,
    int* __restrict__ cnt, int* __restrict__ toks, float* __restrict__ wgts)
{
    const int wave = threadIdx.x >> 5;
    const int lane = threadIdx.x & 31;
    const int t = blockIdx.x * 8 + wave;
    if (t >= T_TOK) return;

    const float* xr = x + (size_t)t * H_DIM;
    float logits[NEXP];
#pragma unroll
    for (int e = 0; e < NEXP; ++e) {
        const float* wr = Wr + (size_t)e * H_DIM;
        float p = 0.f;
        for (int i = lane; i < H_DIM; i += 32)
            p = fmaf(xr[i], wr[i], p);
#pragma unroll
        for (int off = 16; off > 0; off >>= 1)
            p += __shfl_xor(p, off, 32);
        logits[e] = p;
    }
    float mx = logits[0];
#pragma unroll
    for (int e = 1; e < NEXP; ++e) mx = fmaxf(mx, logits[e]);
    float pr[NEXP], sum = 0.f;
#pragma unroll
    for (int e = 0; e < NEXP; ++e) { pr[e] = __expf(logits[e] - mx); sum += pr[e]; }
    const float inv = fast_rcp(sum);
#pragma unroll
    for (int e = 0; e < NEXP; ++e) pr[e] *= inv;

    int e0 = 0;
#pragma unroll
    for (int e = 1; e < NEXP; ++e) if (pr[e] > pr[e0]) e0 = e;
    int e1 = (e0 == 0) ? 1 : 0;
#pragma unroll
    for (int e = 0; e < NEXP; ++e) if (e != e0 && pr[e] > pr[e1]) e1 = e;
    const float s = fast_rcp(pr[e0] + pr[e1]);

    if (lane == 0) {
        int s0 = atomicAdd(&cnt[e0], 1);
        toks[e0 * T_TOK + s0] = t;  wgts[e0 * T_TOK + s0] = pr[e0] * s;
        int s1 = atomicAdd(&cnt[e1], 1);
        toks[e1 * T_TOK + s1] = t;  wgts[e1 * T_TOK + s1] = pr[e1] * s;
    }
}

// ---------------------------------------------------------------------------
// Fused expert MLP: block = (expert, 48 gathered tokens).
//   out[tok] += c * ( silu(h Wg^T) * (h Wu^T) ) Wd^T
// WT = __bf16 (preconverted weights/activations in ws) or float (cvt on the fly).
// ---------------------------------------------------------------------------
template<typename WT>
__global__ __launch_bounds__(NTHREADS) void moe_mlp_kernel(
    const WT* __restrict__ xin,
    const WT* __restrict__ WgW, const WT* __restrict__ WuW, const WT* __restrict__ WdW,
    const int* __restrict__ cnt, const int* __restrict__ toks,
    const float* __restrict__ wgts, float* __restrict__ out)
{
    __shared__ __bf16 hTile[MTILE][HS];     // 48*1032*2 =  99.0 KB
    __shared__ __bf16 act[MTILE][FS];       // 48*264*2  =  24.8 KB
    __shared__ float  cw[MTILE];
    __shared__ int    tk[MTILE];

    const int e    = blockIdx.y;
    const int nTok = cnt[e];
    const int tile = blockIdx.x;
    if (tile * MTILE >= nTok) return;

    const int tid    = threadIdx.x;
    const int wave   = tid >> 5;
    const int lane   = tid & 31;
    const int lane16 = lane & 15;
    const int mrow   = (lane >> 4) * 8;      // C/D frag: elem r -> row mrow+r

    const WT* wg = WgW + (size_t)e * F_DIM * H_DIM;
    const WT* wu = WuW + (size_t)e * F_DIM * H_DIM;
    const WT* wd = WdW + (size_t)e * H_DIM * F_DIM;

    if (tid < MTILE) {
        const int idx = tile * MTILE + tid;
        if (idx < nTok) { tk[tid] = toks[e * T_TOK + idx]; cw[tid] = wgts[e * T_TOK + idx]; }
        else            { tk[tid] = -1;                    cw[tid] = 0.f; }
    }
    __syncthreads();

    // ---- stage h tile into LDS (bf16); 12 x 8-element chunks per thread ----
    {
#pragma unroll
        for (int i = 0; i < (MTILE * H_DIM / 8) / NTHREADS; ++i) {   // 12
            const int chunk = tid + i * NTHREADS;
            const int m   = chunk >> 7;          // 128 chunks per row
            const int c8  = (chunk & 127) * 8;
            const int row = tk[m] < 0 ? 0 : tk[m];
            if constexpr (std::is_same<WT, __bf16>::value) {
#ifdef USE_ASYNC_LDS
                __builtin_amdgcn_global_load_async_to_lds_b128(
                    (g_v4i*)(xin + (size_t)row * H_DIM + c8),
                    (l_v4i*)&hTile[m][c8], 0, 0);
#else
                *(v8bf*)&hTile[m][c8] = *(const v8bf*)(xin + (size_t)row * H_DIM + c8);
#endif
            } else {
                const float4* src = (const float4*)(xin + (size_t)row * H_DIM + c8);
                float4 v0 = src[0], v1 = src[1];
                hTile[m][c8 + 0] = (__bf16)v0.x;  hTile[m][c8 + 1] = (__bf16)v0.y;
                hTile[m][c8 + 2] = (__bf16)v0.z;  hTile[m][c8 + 3] = (__bf16)v0.w;
                hTile[m][c8 + 4] = (__bf16)v1.x;  hTile[m][c8 + 5] = (__bf16)v1.y;
                hTile[m][c8 + 6] = (__bf16)v1.z;  hTile[m][c8 + 7] = (__bf16)v1.w;
            }
        }
#ifdef USE_ASYNC_LDS
        if constexpr (std::is_same<WT, __bf16>::value)
            __builtin_amdgcn_s_wait_asynccnt(0);
#endif
    }
    __syncthreads();

    const v8f zeroV = {0.f, 0.f, 0.f, 0.f, 0.f, 0.f, 0.f, 0.f};
    v8f cacc[MT_N][4];                       // out[48 x 64] per wave, persistent
#pragma unroll
    for (int mt = 0; mt < MT_N; ++mt)
#pragma unroll
        for (int nt = 0; nt < 4; ++nt) cacc[mt][nt] = zeroV;

    const int h0w = wave * (H_DIM / NWAVES); // 64 output columns per wave

    for (int chunk = 0; chunk < F_DIM / FB; ++chunk) {
        const int f0  = chunk * FB;
        const int n0w = f0 + wave * (FB / NWAVES);   // 16 gate/up columns per wave

        // ---- phase 1: g,u[48 x 16] over K=H; B frags reused across 3 M-tiles ----
        v8f cg[MT_N], cu[MT_N];
#pragma unroll
        for (int mt = 0; mt < MT_N; ++mt) { cg[mt] = zeroV; cu[mt] = zeroV; }

        for (int k0 = 0; k0 < H_DIM; k0 += 32) {
            v16bf a[MT_N];
#pragma unroll
            for (int mt = 0; mt < MT_N; ++mt)
                a[mt] = load_a_frag_lds(&hTile[mt * 16][0], HS, k0, lane);
            v16bf bg = load_b_frag(wg, H_DIM, n0w, k0, lane);
            v16bf bu = load_b_frag(wu, H_DIM, n0w, k0, lane);
#pragma unroll
            for (int mt = 0; mt < MT_N; ++mt) {
                cg[mt] = __builtin_amdgcn_wmma_f32_16x16x32_bf16(
                    false, a[mt], false, bg, (short)0, cg[mt], false, false);
                cu[mt] = __builtin_amdgcn_wmma_f32_16x16x32_bf16(
                    false, a[mt], false, bu, (short)0, cu[mt], false, false);
            }
        }

        __syncthreads();   // WAR: previous act chunk fully consumed

        // silu(g) * u * combine-weight -> bf16 act chunk
        const int colb = (n0w - f0) + lane16;
#pragma unroll
        for (int mt = 0; mt < MT_N; ++mt) {
#pragma unroll
            for (int r = 0; r < 8; ++r) {
                const int m = mt * 16 + mrow + r;
                const float g = cg[mt][r];
                const float u = cu[mt][r];
                const float sg = g * fast_rcp(1.f + __expf(-g));
                act[m][colb] = (__bf16)(sg * u * cw[m]);
            }
        }
        __syncthreads();

        if (chunk + 1 < F_DIM / FB)   // pull next Wd chunk toward L2
            __builtin_prefetch(wd + (size_t)(h0w + lane16) * F_DIM + (f0 + FB), 0, 1);

        // ---- phase 2: out[48 x 64] += act_chunk @ Wd[:,chunk]^T ----
        for (int ks = 0; ks < FB; ks += 32) {
            v16bf a[MT_N];
#pragma unroll
            for (int mt = 0; mt < MT_N; ++mt)
                a[mt] = load_a_frag_lds(&act[mt * 16][0], FS, ks, lane);
#pragma unroll
            for (int nt = 0; nt < 4; ++nt) {
                v16bf bd = load_b_frag(wd, F_DIM, h0w + nt * 16, f0 + ks, lane);
#pragma unroll
                for (int mt = 0; mt < MT_N; ++mt)
                    cacc[mt][nt] = __builtin_amdgcn_wmma_f32_16x16x32_bf16(
                        false, a[mt], false, bd, (short)0, cacc[mt][nt], false, false);
            }
        }
    }

    // Scatter-accumulate (token belongs to 2 experts -> fp32 atomics).
#pragma unroll
    for (int nt = 0; nt < 4; ++nt) {
        const int colh = h0w + nt * 16 + lane16;
#pragma unroll
        for (int mt = 0; mt < MT_N; ++mt) {
#pragma unroll
            for (int r = 0; r < 8; ++r) {
                const int m = mt * 16 + mrow + r;
                const int t = tk[m];
                if (t >= 0)
                    atomicAdd(&out[(size_t)t * H_DIM + colh], cacc[mt][nt][r]);
            }
        }
    }
}

// ---------------------------------------------------------------------------
// Launcher. inputs: x, Wr, Wg, Wu, Wd (all f32), topk.
// ws: [0,256) counters | toks E*T*4 | wgts E*T*4 | xbf | WgB | WuB | WdB (bf16)
// ---------------------------------------------------------------------------
extern "C" void kernel_launch(void* const* d_in, const int* in_sizes, int n_in,
                              void* d_out, int out_size, void* d_ws, size_t ws_size,
                              hipStream_t stream) {
    (void)in_sizes; (void)n_in;

    const float* x  = (const float*)d_in[0];
    const float* Wr = (const float*)d_in[1];
    const float* Wg = (const float*)d_in[2];
    const float* Wu = (const float*)d_in[3];
    const float* Wd = (const float*)d_in[4];
    float* out = (float*)d_out;

    const size_t ET4  = (size_t)NEXP * T_TOK * sizeof(int);            // 256 KB
    const size_t XBF  = (size_t)T_TOK * H_DIM * sizeof(__bf16);        // 16 MB
    const size_t WBF  = (size_t)NEXP * F_DIM * H_DIM * sizeof(__bf16); // 64 MB
    char* ws    = (char*)d_ws;
    int*   cnt  = (int*)ws;
    int*   toks = (int*)(ws + 256);
    float* wgts = (float*)(ws + 256 + ET4);
    char*  bulk = ws + 256 + 2 * ET4;
    __bf16* xbf = (__bf16*)(bulk);
    __bf16* WgB = (__bf16*)(bulk + XBF);
    __bf16* WuB = (__bf16*)(bulk + XBF + WBF);
    __bf16* WdB = (__bf16*)(bulk + XBF + 2 * WBF);
    const size_t need = 256 + 2 * ET4 + XBF + 3 * WBF;                 // ~219 MB

    (void)hipMemsetAsync(cnt, 0, 256, stream);
    (void)hipMemsetAsync(out, 0, (size_t)out_size * sizeof(float), stream);

    moe_router_kernel<<<T_TOK / 8, 256, 0, stream>>>(x, Wr, cnt, toks, wgts);

    dim3 grid((T_TOK + MTILE - 1) / MTILE, NEXP);   // expert-major -> L2 weight reuse
    if (ws_size >= need) {
        // one-shot bf16 conversion: halves weight bytes, removes cvt from hot loop
        cvt_f32_bf16_kernel<<<2048, 256, 0, stream>>>(x,  xbf, (long long)T_TOK * H_DIM);
        cvt_f32_bf16_kernel<<<4096, 256, 0, stream>>>(Wg, WgB, (long long)NEXP * F_DIM * H_DIM);
        cvt_f32_bf16_kernel<<<4096, 256, 0, stream>>>(Wu, WuB, (long long)NEXP * F_DIM * H_DIM);
        cvt_f32_bf16_kernel<<<4096, 256, 0, stream>>>(Wd, WdB, (long long)NEXP * H_DIM * F_DIM);
        moe_mlp_kernel<__bf16><<<grid, NTHREADS, 0, stream>>>(
            xbf, WgB, WuB, WdB, cnt, toks, wgts, out);
    } else {
        moe_mlp_kernel<float><<<grid, NTHREADS, 0, stream>>>(
            x, Wg, Wu, Wd, cnt, toks, wgts, out);
    }
}
